// localMPNNet_83562883711517
// MI455X (gfx1250) — compile-verified, matrix operand order checked by reading, not run on hardware
//
#include <hip/hip_runtime.h>
#include <math.h>

// ---------------------------------------------------------------------------
// MPNN (NNConv + GRU x3 + Set2Set x3) for MI455X / gfx1250, wave32 + WMMA.
//
// Heavy math (edge message GEMM, out@root, GRU input/hidden GEMMs) runs on
// v_wmma_f32_16x16x32_bf16.  The edge-weight tensor W_e (1.6 GB) is never
// materialized: msg[e,o] = sum_{i,h} out[src,i]*h_e[e,h]*M[h,i,o] is computed
// as a GEMM whose A rows are per-edge rank-1 products built on the fly from
// LDS (K = 64*128 + 64 bias rows = 8256), against a fixed repacked B
// (8256x64 bf16 ~= 1 MB, L2-resident).  Scatter-add into agg is fused in.
// B tiles are double-buffered (software pipelined) so WMMA never waits on
// same-iteration loads; wave-id is readfirstlane'd so control flow is scalar.
// Workspace need: ~142 MB.
// ---------------------------------------------------------------------------

constexpr int Ni  = 50000;   // nodes
constexpr int Ei  = 100000;  // edges
constexpr int BG  = 2000;    // graphs
constexpr int NPG = 25;      // nodes per graph (N/B, sorted batch)

typedef __attribute__((ext_vector_type(16))) __bf16          v16bf;
typedef __attribute__((ext_vector_type(8)))  float           v8f;

struct alignas(16) Frag { unsigned u[8]; };   // 16 bf16 = one WMMA A/B operand

__device__ __forceinline__ unsigned short f2bf(float f) {
  union { float f; unsigned u; } v; v.f = f;
  unsigned r = v.u + 0x7fffu + ((v.u >> 16) & 1u);   // RNE
  return (unsigned short)(r >> 16);
}
__device__ __forceinline__ float bf2f(unsigned short b) {
  union { unsigned u; float f; } v; v.u = ((unsigned)b) << 16; return v.f;
}
// pack hi16(f0), hi16(f1) -> dword (bf16 truncation) with one v_perm_b32
__device__ __forceinline__ unsigned pk_bf_trunc(float f0, float f1) {
  return __builtin_amdgcn_perm(__builtin_bit_cast(unsigned, f1),
                               __builtin_bit_cast(unsigned, f0), 0x07060302u);
}
__device__ __forceinline__ float sigm(float x) { return 1.f / (1.f + expf(-x)); }

__device__ __forceinline__ v8f wmma_bf(const Frag& a, const Frag& b, v8f c) {
  return __builtin_amdgcn_wmma_f32_16x16x32_bf16(
      false, __builtin_bit_cast(v16bf, a),
      false, __builtin_bit_cast(v16bf, b),
      (short)0, c, false, false);
}

// k-offset inside a 32-wide K tile for A/B fragment dword j, half kh (lane>>4)
__device__ __forceinline__ int koff(int j, int kh) {
  return (j < 4) ? (kh * 8 + 2 * j) : (16 + kh * 8 + 2 * (j - 4));
}

// ----------------------------- workspace layout ----------------------------
constexpr size_t alup(size_t x) { return (x + 255) & ~(size_t)255; }
constexpr size_t OFF_HE    = 0;                                      // [E,128] bf16
constexpr size_t OFF_HBF   = OFF_HE    + alup((size_t)Ei * 128 * 2); // out/h bf16 [N,64]
constexpr size_t OFF_HST   = OFF_HBF   + alup((size_t)Ni * 64 * 2);  // h f32 [N,64]
constexpr size_t OFF_AGG   = OFF_HST   + alup((size_t)Ni * 64 * 4);  // agg f32 [N,64]
constexpr size_t OFF_MBF   = OFF_AGG   + alup((size_t)Ni * 64 * 4);  // m bf16 [N,64]
constexpr size_t OFF_GI    = OFF_MBF   + alup((size_t)Ni * 64 * 2);  // gi f32 [N,192]
constexpr size_t OFF_GH    = OFF_GI    + alup((size_t)Ni * 192 * 4); // gh f32 [N,192]
constexpr size_t OFF_DEG   = OFF_GH    + alup((size_t)Ni * 192 * 4); // deg f32 [N]
constexpr size_t OFF_B2P   = OFF_DEG   + alup((size_t)Ni * 4);       // packed B2 258kt*4nt
constexpr size_t OFF_ROOTP = OFF_B2P   + alup((size_t)258 * 1024 * 4);
constexpr size_t OFF_WIHP  = OFF_ROOTP + alup((size_t)2 * 4 * 32 * 8 * 4);
constexpr size_t OFF_WHHP  = OFF_WIHP  + alup((size_t)2 * 12 * 32 * 8 * 4);
constexpr size_t OFF_QH    = OFF_WHHP  + alup((size_t)2 * 12 * 32 * 8 * 4);
constexpr size_t OFF_QC    = OFF_QH    + alup((size_t)BG * 64 * 4);
constexpr size_t OFF_QST   = OFF_QC    + alup((size_t)BG * 64 * 4);
constexpr size_t OFF_EA    = OFF_QST   + alup((size_t)BG * 128 * 4);
constexpr size_t WS_NEED   = OFF_EA    + alup((size_t)Ni * 4);       // ~142 MB

// ----------------------------- weight packing ------------------------------
// Pack B[k][n] (bf16) into per-(ktile,ntile) wave32 WMMA fragments:
// lane(0-15)=column, lane>>4 picks K half, 8 dwords (2 bf16 each) per lane.
__global__ void k_pack_b(const float* __restrict__ src, unsigned* __restrict__ dst,
                         int K, int N, int transpose) {
  int tid = blockIdx.x * blockDim.x + threadIdx.x;
  int KT = K / 32, NT = N / 16;
  if (tid >= KT * NT * 32) return;
  int lane = tid & 31, tile = tid >> 5;
  int nt = tile % NT, kt = tile / NT;
  int col = nt * 16 + (lane & 15);
  int kb  = kt * 32;
  unsigned* o = dst + (size_t)tile * 32 * 8 + lane * 8;
  for (int j = 0; j < 8; ++j) {
    int k0 = kb + koff(j, lane >> 4), k1 = k0 + 1;
    float b0 = transpose ? src[col * K + k0] : src[k0 * N + col];
    float b1 = transpose ? src[col * K + k1] : src[k1 * N + col];
    o[j] = (unsigned)f2bf(b0) | ((unsigned)f2bf(b1) << 16);
  }
}

// B2[k][o]: k<8192 -> mlp_w2[(i*64+o)*128 + h], i=k>>7, h=k&127
//           k>=8192 -> mlp_b2[(k-8192)*64 + o]  (bias folded as extra K rows)
__device__ __forceinline__ float b2elem(int k, int o, const float* w2, const float* b2) {
  if (k < 8192) { int i = k >> 7, h = k & 127; return w2[(i * 64 + o) * 128 + h]; }
  return b2[(k - 8192) * 64 + o];
}
__global__ void k_pack_b2(const float* __restrict__ w2, const float* __restrict__ b2,
                          unsigned* __restrict__ dst) {
  int tid = blockIdx.x * blockDim.x + threadIdx.x;
  if (tid >= 258 * 4 * 32) return;
  int lane = tid & 31, tile = tid >> 5;
  int nt = tile & 3, kt = tile >> 2;
  int col = nt * 16 + (lane & 15);
  unsigned* o = dst + (size_t)tile * 32 * 8 + lane * 8;
  for (int j = 0; j < 8; ++j) {
    int k0 = kt * 32 + koff(j, lane >> 4);
    o[j] = (unsigned)f2bf(b2elem(k0, col, w2, b2)) |
           ((unsigned)f2bf(b2elem(k0 + 1, col, w2, b2)) << 16);
  }
}

// ------------------------------ prologue -----------------------------------
__global__ __launch_bounds__(256) void k_lin0(const float* __restrict__ x,
    const float* __restrict__ w, const float* __restrict__ b,
    float* __restrict__ h, unsigned short* __restrict__ hbf) {
  int t = blockIdx.x * 256 + threadIdx.x;
  if (t >= Ni * 64) return;
  int n = t >> 6, d = t & 63;
  float s = b[d];
  for (int f = 0; f < 14; ++f) s += x[n * 14 + f] * w[d * 14 + f];
  s = fmaxf(s, 0.f);
  h[t] = s; hbf[t] = f2bf(s);
}

__global__ __launch_bounds__(256) void k_edge_mlp1(const float* __restrict__ ea,
    const float* __restrict__ w1, const float* __restrict__ b1,
    unsigned short* __restrict__ he) {
  long long t = (long long)blockIdx.x * 256 + threadIdx.x;
  if (t >= (long long)Ei * 128) return;
  int e = (int)(t >> 7), hh = (int)(t & 127);
  float s = b1[hh];
  for (int f = 0; f < 4; ++f) s += ea[e * 4 + f] * w1[hh * 4 + f];
  he[t] = f2bf(fmaxf(s, 0.f));
}

__global__ void k_deg(const int* __restrict__ ei, float* __restrict__ deg) {
  int e = blockIdx.x * 256 + threadIdx.x;
  if (e >= Ei) return;
  atomicAdd(&deg[ei[Ei + e]], 1.f);
}

// ------------- fused edge-message GEMM + scatter-add (the big one) ---------
__global__ __launch_bounds__(256) void k_msg(const int* __restrict__ ei,
    const unsigned short* __restrict__ hbf, const unsigned short* __restrict__ he,
    const unsigned* __restrict__ b2p, float* __restrict__ agg) {
  __shared__ int src_s[16], dst_s[16];
  __shared__ unsigned short out_s[16 * 64];
  __shared__ unsigned short he_s[16 * 128];
  __shared__ float cred[16 * 64];
  int t = threadIdx.x, e0 = blockIdx.x * 16;
  if (t < 16) { src_s[t] = ei[e0 + t]; dst_s[t] = ei[Ei + e0 + t]; }
  __syncthreads();
  for (int i = t; i < 1024; i += 256) {
    out_s[i] = hbf[(size_t)src_s[i >> 6] * 64 + (i & 63)];
    cred[i] = 0.f;
  }
  for (int i = t; i < 2048; i += 256)
    he_s[i] = he[(size_t)(e0 + (i >> 7)) * 128 + (i & 127)];
  __syncthreads();

  int wave = __builtin_amdgcn_readfirstlane((int)(t >> 5));  // wave-uniform
  int lane = t & 31;
  int mloc = lane & 15, kh = lane >> 4;
  const unsigned* bp_lane = b2p + (size_t)lane * 8;

  // load the 4 N-tile B fragments of K-tile kt
  auto loadB = [&](Frag* B, int kt) {
#pragma unroll
    for (int nt = 0; nt < 4; ++nt) {
      const uint4* q = (const uint4*)(bp_lane + (size_t)(kt * 4 + nt) * 256);
      *(uint4*)&B[nt].u[0] = q[0];
      *(uint4*)&B[nt].u[4] = q[1];
    }
  };
  // A fragment for K-tile kt<256: rank-1 product out[src,i] * h_e[h]
  auto buildA = [&](int kt) {
    Frag A;
    int iidx = kt >> 2, hbase = (kt & 3) * 32;
    float av = bf2f(out_s[mloc * 64 + iidx]);
#pragma unroll
    for (int j = 0; j < 8; ++j) {
      unsigned p = *(const unsigned*)&he_s[mloc * 128 + hbase + koff(j, kh)];
      float f0 = av * bf2f((unsigned short)(p & 0xffffu));
      float f1 = av * bf2f((unsigned short)(p >> 16));
      A.u[j] = pk_bf_trunc(f0, f1);
    }
    return A;
  };

  v8f acc[4] = {};
  Frag B0[4], B1[4];
  loadB(B0, wave);
  // main K loop over 256 product tiles, unroll x2 with double-buffered B
  for (int kt = wave; kt < 256; kt += 16) {
    loadB(B1, kt + 8);                       // always < 256 inside loop
    {
      Frag A = buildA(kt);
#pragma unroll
      for (int nt = 0; nt < 4; ++nt) acc[nt] = wmma_bf(A, B0[nt], acc[nt]);
    }
    int ktn = kt + 16; if (ktn > 257) ktn = 256 + wave > 257 ? 257 : 256 + wave;
    if (ktn > 257) ktn = 257;
    loadB(B0, ktn);                          // last iter prefetches bias tile
    {
      Frag A = buildA(kt + 8);
#pragma unroll
      for (int nt = 0; nt < 4; ++nt) acc[nt] = wmma_bf(A, B1[nt], acc[nt]);
    }
  }
  // bias tail: K-tiles 256 (i=0..31) and 257 (i=32..63); A = out rows verbatim.
  // After the loop, B0 already holds tile 256+wave for waves 0 and 1.
  if (wave < 2) {
    int ib = wave * 32;
    Frag A;
#pragma unroll
    for (int j = 0; j < 8; ++j)
      A.u[j] = *(const unsigned*)&out_s[mloc * 64 + ib + koff(j, kh)];
#pragma unroll
    for (int nt = 0; nt < 4; ++nt) acc[nt] = wmma_bf(A, B0[nt], acc[nt]);
  }

  // reduce the 8 wave-partials in LDS, then scatter-add per destination node
#pragma unroll
  for (int nt = 0; nt < 4; ++nt)
#pragma unroll
    for (int r = 0; r < 8; ++r)
      atomicAdd(&cred[(r + kh * 8) * 64 + nt * 16 + mloc], acc[nt][r]);
  __syncthreads();
  for (int i = t; i < 1024; i += 256)
    atomicAdd(&agg[(size_t)dst_s[i >> 6] * 64 + (i & 63)], cred[i]);
}

// ---------------- m = relu(agg/deg + out@root + conv_b) --------------------
__global__ __launch_bounds__(256) void k_node(const unsigned short* __restrict__ hbf,
    const unsigned* __restrict__ rootp, const float* __restrict__ agg,
    const float* __restrict__ deg, const float* __restrict__ convb,
    unsigned short* __restrict__ mbf) {
  int wave = __builtin_amdgcn_readfirstlane((int)(threadIdx.x >> 5));
  int lane = threadIdx.x & 31;
  int tile = blockIdx.x * 8 + wave;
  if (tile >= Ni / 16) return;
  int mloc = lane & 15, kh = lane >> 4;
  int row_a = tile * 16 + mloc;
  v8f acc[4] = {};
#pragma unroll
  for (int kt = 0; kt < 2; ++kt) {
    Frag A;
#pragma unroll
    for (int j = 0; j < 8; ++j)
      A.u[j] = *(const unsigned*)&hbf[(size_t)row_a * 64 + kt * 32 + koff(j, kh)];
#pragma unroll
    for (int nt = 0; nt < 4; ++nt) {
      Frag B;
      const uint4* q = (const uint4*)(rootp + ((size_t)(kt * 4 + nt) * 32 + lane) * 8);
      *(uint4*)&B.u[0] = q[0]; *(uint4*)&B.u[4] = q[1];
      acc[nt] = wmma_bf(A, B, acc[nt]);
    }
  }
#pragma unroll
  for (int nt = 0; nt < 4; ++nt)
#pragma unroll
    for (int r = 0; r < 8; ++r) {
      int row = tile * 16 + r + kh * 8, col = nt * 16 + mloc;
      float d = fmaxf(deg[row], 1.f);
      float v = acc[nt][r] + agg[(size_t)row * 64 + col] / d + convb[col];
      mbf[(size_t)row * 64 + col] = f2bf(fmaxf(v, 0.f));
    }
}

// ---------------- C[N,192] = A_bf16[N,64] @ Bpacked + bias -----------------
__global__ __launch_bounds__(256) void k_gates(const unsigned short* __restrict__ abf,
    const unsigned* __restrict__ bp, const float* __restrict__ bias,
    float* __restrict__ outp) {
  int wave = __builtin_amdgcn_readfirstlane((int)(threadIdx.x >> 5));
  int lane = threadIdx.x & 31;
  int task = blockIdx.x * 8 + wave;
  if (task >= (Ni / 16) * 3) return;
  int tile = task / 3, grp = task % 3;   // grp selects 64 of the 192 columns
  int mloc = lane & 15, kh = lane >> 4;
  int row_a = tile * 16 + mloc;
  v8f acc[4] = {};
#pragma unroll
  for (int kt = 0; kt < 2; ++kt) {
    Frag A;
#pragma unroll
    for (int j = 0; j < 8; ++j)
      A.u[j] = *(const unsigned*)&abf[(size_t)row_a * 64 + kt * 32 + koff(j, kh)];
#pragma unroll
    for (int nt = 0; nt < 4; ++nt) {
      Frag B;
      int bt = kt * 12 + grp * 4 + nt;
      const uint4* q = (const uint4*)(bp + ((size_t)bt * 32 + lane) * 8);
      *(uint4*)&B.u[0] = q[0]; *(uint4*)&B.u[4] = q[1];
      acc[nt] = wmma_bf(A, B, acc[nt]);
    }
  }
#pragma unroll
  for (int nt = 0; nt < 4; ++nt)
#pragma unroll
    for (int r = 0; r < 8; ++r) {
      int row = tile * 16 + r + kh * 8;
      int col = grp * 64 + nt * 16 + mloc;
      outp[(size_t)row * 192 + col] = acc[nt][r] + bias[col];
    }
}

__global__ __launch_bounds__(256) void k_gru(const float* __restrict__ gi,
    const float* __restrict__ gh, float* __restrict__ h,
    unsigned short* __restrict__ hbf) {
  int t = blockIdx.x * 256 + threadIdx.x;
  if (t >= Ni * 64) return;
  int n = t >> 6, d = t & 63;
  float ir = gi[(size_t)n * 192 + d],        hr = gh[(size_t)n * 192 + d];
  float iz = gi[(size_t)n * 192 + 64 + d],   hz = gh[(size_t)n * 192 + 64 + d];
  float in_ = gi[(size_t)n * 192 + 128 + d], hn = gh[(size_t)n * 192 + 128 + d];
  float r = sigm(ir + hr), z = sigm(iz + hz);
  float ng = tanhf(in_ + r * hn);
  float hnew = (1.f - z) * ng + z * h[t];
  h[t] = hnew; hbf[t] = f2bf(hnew);
}

// ------------------------------ Set2Set ------------------------------------
__global__ __launch_bounds__(256) void k_lstm(const float* __restrict__ qst,
    float* __restrict__ qh, float* __restrict__ qc,
    const float* __restrict__ wih, const float* __restrict__ whh,
    const float* __restrict__ bih, const float* __restrict__ bhh) {
  int g = blockIdx.x, j = threadIdx.x;
  __shared__ float qs_s[128], qh_s[64], gate_s[256];
  if (j < 128) qs_s[j] = qst[(size_t)g * 128 + j];
  if (j < 64)  qh_s[j] = qh[(size_t)g * 64 + j];
  __syncthreads();
  float s = bih[j] + bhh[j];
  for (int k = 0; k < 128; ++k) s += qs_s[k] * wih[j * 128 + k];
  for (int k = 0; k < 64;  ++k) s += qh_s[k] * whh[j * 64 + k];
  gate_s[j] = s;
  __syncthreads();
  if (j < 64) {
    float i = sigm(gate_s[j]), f = sigm(gate_s[64 + j]);
    float gg = tanhf(gate_s[128 + j]), o = sigm(gate_s[192 + j]);
    float c = f * qc[(size_t)g * 64 + j] + i * gg;
    qc[(size_t)g * 64 + j] = c;
    qh[(size_t)g * 64 + j] = o * tanhf(c);
  }
}

__global__ __launch_bounds__(256) void k_attn(const float* __restrict__ h,
    const float* __restrict__ qh, const int* __restrict__ batch,
    float* __restrict__ e) {
  int wave = threadIdx.x >> 5, lane = threadIdx.x & 31;
  int n = blockIdx.x * 8 + wave;
  if (n >= Ni) return;
  int g = batch[n];
  float s = h[(size_t)n * 64 + lane]      * qh[(size_t)g * 64 + lane] +
            h[(size_t)n * 64 + 32 + lane] * qh[(size_t)g * 64 + 32 + lane];
  for (int off = 16; off; off >>= 1) s += __shfl_xor(s, off, 32);
  if (lane == 0) e[n] = s;
}

__global__ void k_pool(const float* __restrict__ h, const float* __restrict__ e,
                       const float* __restrict__ qh, float* __restrict__ qst) {
  int g = blockIdx.x, lane = threadIdx.x;   // 32 threads
  __shared__ float a_s[NPG];
  int base = g * NPG;
  float ev = (lane < NPG) ? e[base + lane] : -3.402823e38f;
  for (int off = 16; off; off >>= 1) ev = fmaxf(ev, __shfl_xor(ev, off, 32));
  float av = (lane < NPG) ? expf(e[base + lane] - ev) : 0.f;
  float sv = av;
  for (int off = 16; off; off >>= 1) sv += __shfl_xor(sv, off, 32);
  if (lane < NPG) a_s[lane] = av / sv;
  __syncthreads();
  float r0 = 0.f, r1 = 0.f;
  for (int k = 0; k < NPG; ++k) {
    float a = a_s[k];
    r0 += a * h[(size_t)(base + k) * 64 + lane];
    r1 += a * h[(size_t)(base + k) * 64 + 32 + lane];
  }
  qst[(size_t)g * 128 + lane]      = qh[(size_t)g * 64 + lane];
  qst[(size_t)g * 128 + 32 + lane] = qh[(size_t)g * 64 + 32 + lane];
  qst[(size_t)g * 128 + 64 + lane] = r0;
  qst[(size_t)g * 128 + 96 + lane] = r1;
}

__global__ void k_readout(const float* __restrict__ qst, const float* __restrict__ w1,
    const float* __restrict__ b1, const float* __restrict__ w2,
    const float* __restrict__ b2, float* __restrict__ y) {
  int g = blockIdx.x, j = threadIdx.x;   // 64 threads
  __shared__ float qs_s[128], part[64];
  qs_s[j] = qst[(size_t)g * 128 + j];
  qs_s[64 + j] = qst[(size_t)g * 128 + 64 + j];
  __syncthreads();
  float s = b1[j];
  for (int k = 0; k < 128; ++k) s += qs_s[k] * w1[j * 128 + k];
  part[j] = fmaxf(s, 0.f) * w2[j];
  __syncthreads();
  if (j == 0) {
    float acc = b2[0];
    for (int k = 0; k < 64; ++k) acc += part[k];
    y[g] = acc;
  }
}

// ------------------------------- driver ------------------------------------
extern "C" void kernel_launch(void* const* d_in, const int* in_sizes, int n_in,
                              void* d_out, int out_size, void* d_ws, size_t ws_size,
                              hipStream_t stream) {
  (void)in_sizes; (void)n_in; (void)out_size; (void)ws_size;
  const float* x     = (const float*)d_in[0];
  const float* ea    = (const float*)d_in[1];
  const int*   ei    = (const int*)d_in[2];
  const int*   batch = (const int*)d_in[3];
  const float* lin0w = (const float*)d_in[4];
  const float* lin0b = (const float*)d_in[5];
  const float* w1    = (const float*)d_in[6];
  const float* b1    = (const float*)d_in[7];
  const float* w2    = (const float*)d_in[8];
  const float* b2    = (const float*)d_in[9];
  const float* root  = (const float*)d_in[10];
  const float* convb = (const float*)d_in[11];
  const float* gwih  = (const float*)d_in[12];
  const float* gwhh  = (const float*)d_in[13];
  const float* gbih  = (const float*)d_in[14];
  const float* gbhh  = (const float*)d_in[15];
  const float* lwih  = (const float*)d_in[16];
  const float* lwhh  = (const float*)d_in[17];
  const float* lbih  = (const float*)d_in[18];
  const float* lbhh  = (const float*)d_in[19];
  const float* l1w   = (const float*)d_in[20];
  const float* l1b   = (const float*)d_in[21];
  const float* l2w   = (const float*)d_in[22];
  const float* l2b   = (const float*)d_in[23];

  char* ws = (char*)d_ws;
  unsigned short* he   = (unsigned short*)(ws + OFF_HE);
  unsigned short* hbf  = (unsigned short*)(ws + OFF_HBF);
  float*          hst  = (float*)(ws + OFF_HST);
  float*          agg  = (float*)(ws + OFF_AGG);
  unsigned short* mbf  = (unsigned short*)(ws + OFF_MBF);
  float*          gi   = (float*)(ws + OFF_GI);
  float*          gh   = (float*)(ws + OFF_GH);
  float*          deg  = (float*)(ws + OFF_DEG);
  unsigned*       b2p  = (unsigned*)(ws + OFF_B2P);
  unsigned*       rootp= (unsigned*)(ws + OFF_ROOTP);
  unsigned*       wihp = (unsigned*)(ws + OFF_WIHP);
  unsigned*       whhp = (unsigned*)(ws + OFF_WHHP);
  float*          qh   = (float*)(ws + OFF_QH);
  float*          qc   = (float*)(ws + OFF_QC);
  float*          qst  = (float*)(ws + OFF_QST);
  float*          eatt = (float*)(ws + OFF_EA);
  float*          y    = (float*)d_out;

  hipMemsetAsync(deg, 0, (size_t)Ni * 4, stream);
  hipMemsetAsync(qh,  0, (size_t)BG * 64 * 4, stream);
  hipMemsetAsync(qc,  0, (size_t)BG * 64 * 4, stream);
  hipMemsetAsync(qst, 0, (size_t)BG * 128 * 4, stream);

  // weight repack (bf16 WMMA fragment layout)
  k_pack_b2<<<(258 * 4 * 32 + 255) / 256, 256, 0, stream>>>(w2, b2, b2p);
  k_pack_b<<<1, 256, 0, stream>>>(root, rootp, 64, 64, 0);
  k_pack_b<<<3, 256, 0, stream>>>(gwih, wihp, 64, 192, 1);
  k_pack_b<<<3, 256, 0, stream>>>(gwhh, whhp, 64, 192, 1);

  // prologue
  k_lin0<<<(Ni * 64 + 255) / 256, 256, 0, stream>>>(x, lin0w, lin0b, hst, hbf);
  k_edge_mlp1<<<(int)(((long long)Ei * 128 + 255) / 256), 256, 0, stream>>>(ea, w1, b1, he);
  k_deg<<<(Ei + 255) / 256, 256, 0, stream>>>(ei, deg);

  // 3 message-passing + GRU steps
  for (int it = 0; it < 3; ++it) {
    hipMemsetAsync(agg, 0, (size_t)Ni * 64 * 4, stream);
    k_msg<<<Ei / 16, 256, 0, stream>>>(ei, hbf, he, b2p, agg);
    k_node<<<(Ni / 16 + 7) / 8, 256, 0, stream>>>(hbf, rootp, agg, deg, convb, mbf);
    k_gates<<<((Ni / 16) * 3 + 7) / 8, 256, 0, stream>>>(mbf, wihp, gbih, gi);
    k_gates<<<((Ni / 16) * 3 + 7) / 8, 256, 0, stream>>>(hbf, whhp, gbhh, gh);
    k_gru<<<(Ni * 64 + 255) / 256, 256, 0, stream>>>(gi, gh, hst, hbf);
  }

  // Set2Set pooling, 3 steps
  for (int s = 0; s < 3; ++s) {
    k_lstm<<<BG, 256, 0, stream>>>(qst, qh, qc, lwih, lwhh, lbih, lbhh);
    k_attn<<<(Ni + 7) / 8, 256, 0, stream>>>(hst, qh, batch, eatt);
    k_pool<<<BG, 32, 0, stream>>>(hst, eatt, qh, qst);
  }
  k_readout<<<BG, 64, 0, stream>>>(qst, l1w, l1b, l2w, l2b, y);
}
